// TemporalGNN_62156766708260
// MI455X (gfx1250) — compile-verified, matrix operand order checked by reading, not run on hardware
//
#include <hip/hip_runtime.h>
#include <hip/hip_bf16.h>
#include <cstdint>
#include <cstddef>

// ---------------------------------------------------------------------------
// Problem constants (fixed by the reference)
// ---------------------------------------------------------------------------
#define NNODES 20000
#define NEDGES 320000
#define TSTEPS 24
#define SDIM 16
#define DDIM 8
#define HID 128
#define HEADS 4
#define DPH 32
#define NLAYERS 3
#define EPLUS (NEDGES + NNODES)   // edges + self loops

typedef __bf16 bf16_t;
typedef __attribute__((ext_vector_type(16))) __bf16 v16bf;
typedef __attribute__((ext_vector_type(8)))  __bf16 v8bf;
typedef __attribute__((ext_vector_type(8)))  float  v8f;

union AFrag { v16bf v; v8bf h[2]; };

__device__ __forceinline__ float sigmoidf_(float x) { return 1.0f / (1.0f + __expf(-x)); }

// Order-preserving float<->uint encoding so atomicMax(u32) == float max.
__device__ __forceinline__ unsigned f32_ord(float x) {
    unsigned u = __float_as_uint(x);
    return (u & 0x80000000u) ? ~u : (u | 0x80000000u);
}
__device__ __forceinline__ float ord_f32(unsigned u) {
    return __uint_as_float((u & 0x80000000u) ? (u ^ 0x80000000u) : ~u);
}

// ---------------------------------------------------------------------------
// Fill (zero-init of scratch; 0u is also the -inf sentinel for the
// order-encoded max buffer and bf16/f32 zero)
// ---------------------------------------------------------------------------
__global__ void fill_u32(unsigned* __restrict__ p, size_t nwords, unsigned v) {
    size_t i = (size_t)blockIdx.x * blockDim.x + threadIdx.x;
    if (i < nwords) p[i] = v;
}

// ---------------------------------------------------------------------------
// Pack a K=128 x NC f32 weight matrix into bf16 WMMA B-fragment layout.
// Fragment tile (tk,tn): per-lane 16 bf16 values stored contiguously (32B),
// mirroring the 16-bit A-matrix K distribution from the CDNA5 ISA tables.
// transpose=1 reads src as [NC,128] (i.e. B[k][n] = src[n*128+k], for W^T).
// ---------------------------------------------------------------------------
__global__ void pack_w(const float* __restrict__ src, bf16_t* __restrict__ dst,
                       int NC, int transpose) {
    int tid = blockIdx.x * blockDim.x + threadIdx.x;
    int total = HID * NC;
    if (tid >= total) return;
    int s    = tid & 15;
    int lane = (tid >> 4) & 31;
    int rest = tid >> 9;
    int ntiles = NC >> 4;
    int tn = rest % ntiles;
    int tk = rest / ntiles;
    int k = 32 * tk + ((s >> 3) << 4) + ((lane >> 4) << 3) + (s & 7);
    int n = 16 * tn + (lane & 15);
    float v = transpose ? src[n * HID + k] : src[k * NC + n];
    dst[tid] = (bf16_t)v;
}

// ---------------------------------------------------------------------------
// Async copy of one B-fragment chunk (4 N-tiles x 4 K-steps = 16 KB) into LDS
// using the gfx1250 async global->LDS data mover (tracked by ASYNCcnt).
// 128 threads x 8 x 16B segments. LDS layout: [kk*4+tnl][1KB frag].
// ---------------------------------------------------------------------------
__device__ __forceinline__ void stage_chunk_async(const bf16_t* __restrict__ B,
                                                  int tn0, int ntiles,
                                                  unsigned ldsByteBase, int t) {
#pragma unroll
    for (int i = 0; i < 8; i++) {
        int seg    = t + i * 128;        // 0..1023 (16B segments)
        int frag   = seg >> 6;           // 0..15
        int kk     = frag >> 2;
        int tnl    = frag & 3;
        int within = seg & 63;           // 16B units inside the 1KB fragment
        const bf16_t* g = B + ((size_t)(kk * ntiles + tn0 + tnl) * 512) + within * 8;
        unsigned l = ldsByteBase + (unsigned)seg * 16u;
        asm volatile("global_load_async_to_lds_b128 %0, %1, off"
                     :: "v"(l), "v"(g) : "memory");
    }
}

// ---------------------------------------------------------------------------
// Generic bf16 WMMA GEMM:  C[M,NC] = A1[M,128] @ B1 (+ A2[M,128] @ B2) + bias
// A row-major bf16, B pre-packed fragments (staged via LDS), C f32. K = 128.
// Block = 4 waves; wave w owns rows [blk*64 + w*16, +16).  All threads join
// the staging barriers (no early return); stores are row-guarded.
// Dynamic LDS: 32 KB (B1 chunk at 0, B2 chunk at 16 KB).
// ---------------------------------------------------------------------------
__global__ void __launch_bounds__(128)
gemm_bf16(const bf16_t* __restrict__ A1, const bf16_t* __restrict__ B1,
          const bf16_t* __restrict__ A2, const bf16_t* __restrict__ B2,
          const float* __restrict__ bias1, const float* __restrict__ bias2,
          float* __restrict__ C, int M, int NC) {
    extern __shared__ bf16_t sB[];           // dynamic LDS, base offset 0
    const int t    = threadIdx.x;
    const int lane = t & 31;
    const int wave = t >> 5;
    const int m0 = (blockIdx.x * 4 + wave) * 16;
    const int lg = lane >> 4;                // K-half lane group
    const int ml = lane & 15;                // row within tile (A) / col (B,C)
    int arow = m0 + ml;
    if (arow >= M) arow = M - 1;             // clamp loads; stores are guarded

    // Cache all four 16x32 A fragments (K = 0..127) in VGPRs.
    AFrag a1[4], a2[4];
#pragma unroll
    for (int kk = 0; kk < 4; kk++) {
        const bf16_t* p = A1 + (size_t)arow * HID + kk * 32 + lg * 8;
        a1[kk].h[0] = *(const v8bf*)(p);
        a1[kk].h[1] = *(const v8bf*)(p + 16);
    }
    if (A2) {
#pragma unroll
        for (int kk = 0; kk < 4; kk++) {
            const bf16_t* p = A2 + (size_t)arow * HID + kk * 32 + lg * 8;
            a2[kk].h[0] = *(const v8bf*)(p);
            a2[kk].h[1] = *(const v8bf*)(p + 16);
        }
    }

    const int ntiles  = NC >> 4;
    const int nchunks = ntiles >> 2;         // 4 N-tiles per chunk
    for (int ch = 0; ch < nchunks; ch++) {
        __syncthreads();                     // protect previous chunk's reads
        stage_chunk_async(B1, ch * 4, ntiles, 0u, t);
        if (A2) stage_chunk_async(B2, ch * 4, ntiles, 16384u, t);
        asm volatile("s_wait_asynccnt 0x0" ::: "memory");
        __syncthreads();                     // all waves' async data visible

#pragma unroll
        for (int tl = 0; tl < 4; tl++) {
            const int tn = ch * 4 + tl;
            const int ncol = tn * 16 + ml;
            float bv = 0.0f;
            if (bias1) bv += bias1[ncol];
            if (bias2) bv += bias2[ncol];
            v8f acc;
#pragma unroll
            for (int i = 0; i < 8; i++) acc[i] = bv;
#pragma unroll
            for (int kk = 0; kk < 4; kk++) {
                v16bf b = *(const v16bf*)(sB + (size_t)(kk * 4 + tl) * 512 + lane * 16);
                acc = __builtin_amdgcn_wmma_f32_16x16x32_bf16(
                    false, a1[kk].v, false, b, (short)0, acc, false, false);
            }
            if (A2) {
#pragma unroll
                for (int kk = 0; kk < 4; kk++) {
                    v16bf b = *(const v16bf*)(sB + 8192 + (size_t)(kk * 4 + tl) * 512 + lane * 16);
                    acc = __builtin_amdgcn_wmma_f32_16x16x32_bf16(
                        false, a2[kk].v, false, b, (short)0, acc, false, false);
                }
            }
            // C/D layout: VGPR i -> row m0 + i + 8*lg, col = tn*16 + (lane&15)
#pragma unroll
            for (int i = 0; i < 8; i++) {
                int r = m0 + i + 8 * lg;
                if (r < M) C[(size_t)r * NC + ncol] = acc[i];
            }
        }
    }
}

// ---------------------------------------------------------------------------
// Input projection:  h = [x_static | x_dynamic_t] @ proj_W + proj_b
// ---------------------------------------------------------------------------
__global__ void proj_kernel(const float* __restrict__ xs, const float* __restrict__ xd,
                            const float* __restrict__ W, const float* __restrict__ b,
                            float* __restrict__ hF, bf16_t* __restrict__ hB) {
    int tid = blockIdx.x * blockDim.x + threadIdx.x;
    if (tid >= NNODES * HID) return;
    int n = tid / HID, j = tid % HID;
    float a = b[j];
#pragma unroll
    for (int k = 0; k < SDIM; k++) a += xs[n * SDIM + k] * W[k * HID + j];
#pragma unroll
    for (int k = 0; k < DDIM; k++) a += xd[n * DDIM + k] * W[(SDIM + k) * HID + j];
    hF[tid] = a;
    hB[tid] = (bf16_t)a;
}

// ---------------------------------------------------------------------------
// Per-head attention logits:  a_src/a_dst[n,h] = <xp[n,h,:], att[h,:]>
// ---------------------------------------------------------------------------
__global__ void attn_score(const float* __restrict__ xp, const float* __restrict__ as_w,
                           const float* __restrict__ ad_w,
                           float* __restrict__ aS, float* __restrict__ aD) {
    int tid = blockIdx.x * blockDim.x + threadIdx.x;
    if (tid >= NNODES * HEADS) return;
    int n = tid >> 2, hd = tid & 3;
    const float* x = xp + (size_t)n * HID + hd * DPH;
    float s = 0.f, d = 0.f;
#pragma unroll
    for (int k = 0; k < DPH; k++) {
        s += x[k] * as_w[hd * DPH + k];
        d += x[k] * ad_w[hd * DPH + k];
    }
    aS[tid] = s;
    aD[tid] = d;
}

__device__ __forceinline__ void edge_sd(const int* ei, int e, int& s, int& d) {
    if (e < NEDGES) { s = ei[e]; d = ei[NEDGES + e]; }
    else            { s = e - NEDGES; d = s; }    // implicit self loops
}

__global__ void edge_max(const int* __restrict__ ei, const float* __restrict__ aS,
                         const float* __restrict__ aD, unsigned* __restrict__ mx) {
    int tid = blockIdx.x * blockDim.x + threadIdx.x;
    if (tid >= EPLUS * HEADS) return;
    int e = tid >> 2, hd = tid & 3;
    int s, d; edge_sd(ei, e, s, d);
    float v = aS[s * HEADS + hd] + aD[d * HEADS + hd];
    v = v > 0.f ? v : 0.2f * v;                   // leaky_relu(0.2)
    atomicMax(&mx[d * HEADS + hd], f32_ord(v));
}

__global__ void edge_denom(const int* __restrict__ ei, const float* __restrict__ aS,
                           const float* __restrict__ aD, const unsigned* __restrict__ mx,
                           float* __restrict__ den) {
    int tid = blockIdx.x * blockDim.x + threadIdx.x;
    if (tid >= EPLUS * HEADS) return;
    int e = tid >> 2, hd = tid & 3;
    int s, d; edge_sd(ei, e, s, d);
    float v = aS[s * HEADS + hd] + aD[d * HEADS + hd];
    v = v > 0.f ? v : 0.2f * v;
    float m = ord_f32(mx[d * HEADS + hd]);
    atomicAdd(&den[d * HEADS + hd], __expf(v - m));
}

__global__ void __launch_bounds__(HID)
edge_scatter(const int* __restrict__ ei, const float* __restrict__ aS,
             const float* __restrict__ aD, const unsigned* __restrict__ mx,
             const float* __restrict__ den, const float* __restrict__ xp,
             float* __restrict__ accum) {
    int e = blockIdx.x;
    int j = threadIdx.x;
    int hd = j >> 5;
    int s, d; edge_sd(ei, e, s, d);
    float v = aS[s * HEADS + hd] + aD[d * HEADS + hd];
    v = v > 0.f ? v : 0.2f * v;
    float m = ord_f32(mx[d * HEADS + hd]);
    float alpha = __expf(v - m) / (den[d * HEADS + hd] + 1e-16f);
    atomicAdd(&accum[(size_t)d * HID + j], xp[(size_t)s * HID + j] * alpha);
}

// ---------------------------------------------------------------------------
// GAT epilogue: + bias + residual, LayerNorm, ReLU.  One block per node.
// ---------------------------------------------------------------------------
__global__ void __launch_bounds__(HID)
gat_post(const float* __restrict__ accum, const float* __restrict__ gb,
         const float* __restrict__ lng, const float* __restrict__ lnb,
         float* __restrict__ hF, bf16_t* __restrict__ hB) {
    __shared__ float red[4];
    int n = blockIdx.x, j = threadIdx.x;
    size_t idx = (size_t)n * HID + j;
    float v = accum[idx] + gb[j] + hF[idx];
    float s = v;
#pragma unroll
    for (int o = 16; o > 0; o >>= 1) s += __shfl_xor(s, o, 32);
    if ((j & 31) == 0) red[j >> 5] = s;
    __syncthreads();
    float mu = (red[0] + red[1] + red[2] + red[3]) * (1.0f / HID);
    __syncthreads();
    float dv = v - mu;
    float q = dv * dv;
#pragma unroll
    for (int o = 16; o > 0; o >>= 1) q += __shfl_xor(q, o, 32);
    if ((j & 31) == 0) red[j >> 5] = q;
    __syncthreads();
    float var = (red[0] + red[1] + red[2] + red[3]) * (1.0f / HID);
    float y = lng[j] * dv * rsqrtf(var + 1e-5f) + lnb[j];
    y = fmaxf(y, 0.0f);
    hF[idx] = y;
    hB[idx] = (bf16_t)y;
}

// ---------------------------------------------------------------------------
// LSTM gates (torch order i,f,g,o).  z is [N, 4*HID] f32 from gemm_bf16.
// ---------------------------------------------------------------------------
__global__ void lstm_gate(const float* __restrict__ z, float* __restrict__ c,
                          bf16_t* __restrict__ hB) {
    int tid = blockIdx.x * blockDim.x + threadIdx.x;
    if (tid >= NNODES * HID) return;
    int n = tid / HID, j = tid % HID;
    const float* zr = z + (size_t)n * 4 * HID;
    float zi = zr[j], zf = zr[HID + j], zg = zr[2 * HID + j], zo = zr[3 * HID + j];
    float cn = sigmoidf_(zf) * c[tid] + sigmoidf_(zi) * tanhf(zg);
    float h = sigmoidf_(zo) * tanhf(cn);
    c[tid] = cn;
    hB[tid] = (bf16_t)h;
}

// ---------------------------------------------------------------------------
// Output MLP
// ---------------------------------------------------------------------------
__global__ void mlp1(const bf16_t* __restrict__ hB, const float* __restrict__ W1,
                     const float* __restrict__ b1, float* __restrict__ t1) {
    int tid = blockIdx.x * blockDim.x + threadIdx.x;
    if (tid >= NNODES * (HID / 2)) return;
    int n = tid / (HID / 2), jj = tid % (HID / 2);
    float a = b1[jj];
    for (int k = 0; k < HID; k++)
        a += (float)hB[(size_t)n * HID + k] * W1[k * (HID / 2) + jj];
    t1[tid] = fmaxf(a, 0.0f);
}

__global__ void mlp2(const float* __restrict__ t1, const float* __restrict__ W2,
                     const float* __restrict__ b2, float* __restrict__ out) {
    int n = blockIdx.x * blockDim.x + threadIdx.x;
    if (n >= NNODES) return;
    float a = b2[0];
    for (int k = 0; k < HID / 2; k++) a += t1[(size_t)n * (HID / 2) + k] * W2[k];
    out[n] = a;
}

// ---------------------------------------------------------------------------
// Host orchestration
// ---------------------------------------------------------------------------
extern "C" void kernel_launch(void* const* d_in, const int* in_sizes, int n_in,
                              void* d_out, int out_size, void* d_ws, size_t ws_size,
                              hipStream_t stream) {
    (void)in_sizes; (void)n_in; (void)out_size; (void)ws_size;
    const float* xs   = (const float*)d_in[0];
    const float* xd   = (const float*)d_in[1];
    const int*   ei   = (const int*)  d_in[2];
    const float* pW   = (const float*)d_in[3];
    const float* pb   = (const float*)d_in[4];
    const float* gatW = (const float*)d_in[5];
    const float* attS = (const float*)d_in[6];
    const float* attD = (const float*)d_in[7];
    const float* gatB = (const float*)d_in[8];
    const float* lnG  = (const float*)d_in[9];
    const float* lnB  = (const float*)d_in[10];
    const float* Wih0 = (const float*)d_in[11];
    const float* Whh0 = (const float*)d_in[12];
    const float* bih0 = (const float*)d_in[13];
    const float* bhh0 = (const float*)d_in[14];
    const float* Wih1 = (const float*)d_in[15];
    const float* Whh1 = (const float*)d_in[16];
    const float* bih1 = (const float*)d_in[17];
    const float* bhh1 = (const float*)d_in[18];
    const float* oW1  = (const float*)d_in[19];
    const float* ob1  = (const float*)d_in[20];
    const float* oW2  = (const float*)d_in[21];
    const float* ob2  = (const float*)d_in[22];
    float* pred = (float*)d_out;

    // ---- workspace carve-up (fixed offsets, 256B aligned; ~115 MB total) ----
    char* base = (char*)d_ws;
    size_t off = 0;
    auto alloc = [&](size_t bytes) -> char* {
        size_t o = (off + 255) & ~(size_t)255;
        off = o + bytes;
        return base + o;
    };
    float*    hF    = (float*)   alloc((size_t)NNODES * HID * 4);
    bf16_t*   hB    = (bf16_t*)  alloc((size_t)NNODES * HID * 2);
    float*    xp    = (float*)   alloc((size_t)NNODES * HID * 4);
    float*    aS    = (float*)   alloc((size_t)NNODES * HEADS * 4);
    float*    aD    = (float*)   alloc((size_t)NNODES * HEADS * 4);
    unsigned* mx    = (unsigned*)alloc((size_t)NNODES * HEADS * 4 * 2); // mx+den
    float*    den   = (float*)(mx + (size_t)NNODES * HEADS);
    float*    accum = (float*)   alloc((size_t)NNODES * HID * 4);
    float*    z     = (float*)   alloc((size_t)NNODES * 4 * HID * 4);
    bf16_t*   h0B   = (bf16_t*)  alloc((size_t)NNODES * HID * 2);
    float*    c0    = (float*)   alloc((size_t)NNODES * HID * 4);
    bf16_t*   h1B   = (bf16_t*)  alloc((size_t)NNODES * HID * 2);
    float*    c1    = (float*)   alloc((size_t)NNODES * HID * 4);
    float*    t1    = (float*)   alloc((size_t)NNODES * (HID / 2) * 4);
    bf16_t*   pGat  = (bf16_t*)  alloc((size_t)NLAYERS * HID * HID * 2);
    bf16_t*   pIh0  = (bf16_t*)  alloc((size_t)HID * 4 * HID * 2);
    bf16_t*   pHh0  = (bf16_t*)  alloc((size_t)HID * 4 * HID * 2);
    bf16_t*   pIh1  = (bf16_t*)  alloc((size_t)HID * 4 * HID * 2);
    bf16_t*   pHh1  = (bf16_t*)  alloc((size_t)HID * 4 * HID * 2);

    const int thr = 256;
    auto fillz = [&](void* p, size_t words) {
        fill_u32<<<(unsigned)((words + thr - 1) / thr), thr, 0, stream>>>(
            (unsigned*)p, words, 0u);
    };

    // ---- pack weights into WMMA fragment layout (bf16) ----
    {
        int tot = HID * HID, blocks = (tot + thr - 1) / thr;
        for (int l = 0; l < NLAYERS; l++)
            pack_w<<<blocks, thr, 0, stream>>>(gatW + (size_t)l * HID * HID,
                                               pGat + (size_t)l * HID * HID, HID, 0);
        int tot2 = HID * 4 * HID, b2 = (tot2 + thr - 1) / thr;
        pack_w<<<b2, thr, 0, stream>>>(Wih0, pIh0, 4 * HID, 1);
        pack_w<<<b2, thr, 0, stream>>>(Whh0, pHh0, 4 * HID, 1);
        pack_w<<<b2, thr, 0, stream>>>(Wih1, pIh1, 4 * HID, 1);
        pack_w<<<b2, thr, 0, stream>>>(Whh1, pHh1, 4 * HID, 1);
    }

    // ---- zero LSTM state ----
    fillz(h0B, (size_t)NNODES * HID / 2);
    fillz(h1B, (size_t)NNODES * HID / 2);
    fillz(c0, (size_t)NNODES * HID);
    fillz(c1, (size_t)NNODES * HID);

    const int gemmBlocks = (NNODES + 63) / 64;
    const size_t gemmLds = 32768;            // B1 chunk (16KB) + B2 chunk (16KB)
    const int totNH = NNODES * HID;
    const int eThreads = EPLUS * HEADS;

    for (int t = 0; t < TSTEPS; t++) {
        proj_kernel<<<(totNH + thr - 1) / thr, thr, 0, stream>>>(
            xs, xd + (size_t)t * NNODES * DDIM, pW, pb, hF, hB);
        for (int l = 0; l < NLAYERS; l++) {
            fillz(accum, (size_t)NNODES * HID);
            fillz(mx, (size_t)NNODES * HEADS * 2);   // covers mx and den
            gemm_bf16<<<gemmBlocks, 128, gemmLds, stream>>>(
                hB, pGat + (size_t)l * HID * HID, nullptr, nullptr,
                nullptr, nullptr, xp, NNODES, HID);
            attn_score<<<(NNODES * HEADS + thr - 1) / thr, thr, 0, stream>>>(
                xp, attS + l * HEADS * DPH, attD + l * HEADS * DPH, aS, aD);
            edge_max<<<(eThreads + thr - 1) / thr, thr, 0, stream>>>(ei, aS, aD, mx);
            edge_denom<<<(eThreads + thr - 1) / thr, thr, 0, stream>>>(ei, aS, aD, mx, den);
            edge_scatter<<<EPLUS, HID, 0, stream>>>(ei, aS, aD, mx, den, xp, accum);
            gat_post<<<NNODES, HID, 0, stream>>>(accum, gatB + l * HID,
                                                 lnG + l * HID, lnB + l * HID, hF, hB);
        }
        // LSTM cell 0:  z = h_gnn @ Wih0^T + h0 @ Whh0^T + bih0 + bhh0
        gemm_bf16<<<gemmBlocks, 128, gemmLds, stream>>>(hB, pIh0, h0B, pHh0,
                                                        bih0, bhh0, z, NNODES, 4 * HID);
        lstm_gate<<<(totNH + thr - 1) / thr, thr, 0, stream>>>(z, c0, h0B);
        // LSTM cell 1
        gemm_bf16<<<gemmBlocks, 128, gemmLds, stream>>>(h0B, pIh1, h1B, pHh1,
                                                        bih1, bhh1, z, NNODES, 4 * HID);
        lstm_gate<<<(totNH + thr - 1) / thr, thr, 0, stream>>>(z, c1, h1B);
    }

    mlp1<<<(NNODES * (HID / 2) + thr - 1) / thr, thr, 0, stream>>>(h1B, oW1, ob1, t1);
    mlp2<<<(NNODES + thr - 1) / thr, thr, 0, stream>>>(t1, oW2, ob2, pred);
}
